// AttentionGuidedMessagePassing_33809982554513
// MI455X (gfx1250) — matrix operand order repaired and checked by reading, not compiled, not run on hardware
//
#include <hip/hip_runtime.h>
#include <hip/hip_bf16.h>
#include <math.h>

// ---------------------------------------------------------------------------
// CDNA5 (gfx1250) implementation: all 64-channel contractions run on
// v_wmma_f32_16x16x32_f16 (wave32, 16x16 tiles, f32 accumulate).
// LDS row strides padded (96->104, 64->72 halves) so fragment b128 loads are
// bank-conflict-free on the 64x4B LDS banks; directional GEMMs use M=32 per
// wave (two A tiles per B fragment) to halve weight-fragment LDS traffic.
// ---------------------------------------------------------------------------

typedef __attribute__((ext_vector_type(16))) _Float16 v16h;
typedef __attribute__((ext_vector_type(8)))  float    v8f;

#define B_   4
#define H_   256
#define W_   256
#define C_   32
#define HID_ 64
#define NLINES_ 1020   // B * 255 lines per direction

#define XT_S  104      // padded stride for 96-channel tiles
#define OT_S  72       // padded stride for 64-channel tiles

static __device__ __forceinline__ v8f wmma_f16(v16h a, v16h b, v8f c) {
  // D = A(16x32 f16) * B(32x16 f16) + C(16x16 f32)
  return __builtin_amdgcn_wmma_f32_16x16x32_f16(false, a, false, b, (short)0, c,
                                                false, false);
}

// A-fragment (16x32 f16) from an LDS [rows][row_stride] f16 matrix.
// Lane L holds row M=L&15; regs 0-3: K = hi*8 + 2r,2r+1 ;
// regs 4-7: K = 16 + hi*8 + 2(r-4),+1.
static __device__ __forceinline__ v16h load_a_frag(const _Float16* base,
                                                   int row_stride, int m_base,
                                                   int k_base) {
  const int lane = threadIdx.x & 31;
  const int m    = m_base + (lane & 15);
  const int hi   = (lane >> 4) & 1;
  union { v16h v; unsigned int u[8]; } r;
#pragma unroll
  for (int j = 0; j < 8; ++j) {
    int k = (j < 4) ? (hi * 8 + 2 * j) : (16 + hi * 8 + 2 * (j - 4));
    r.u[j] = *(const unsigned int*)(base + (size_t)m * row_stride + k_base + k);
  }
  return r.v;
}

// B-fragment (32x16 f16) from a weight matrix stored [n][k_stride].
// Lane L holds column N=L&15; halves 0-7: K = hi*8 + j ; halves 8-15: K = 16 + hi*8 + (j-8).
static __device__ __forceinline__ v16h load_b_frag(const _Float16* w,
                                                   int k_stride, int n_base,
                                                   int k_base) {
  const int lane = threadIdx.x & 31;
  const int n    = n_base + (lane & 15);
  const int hi   = (lane >> 4) & 1;
  const _Float16* p = w + (size_t)n * k_stride + k_base + hi * 8;
  union { v16h v; uint4 q[2]; } r;
  r.q[0] = *(const uint4*)(p);
  r.q[1] = *(const uint4*)(p + 16);
  return r.v;
}

// ---------------------------------------------------------------------------
// Weight packing: f32 OIHW slices -> f16 [n][k] B matrices (K padded to 32x).
// ---------------------------------------------------------------------------
__global__ void k_pack_w(const float* __restrict__ aw1, const float* __restrict__ gw1,
                         const float* __restrict__ gw2, const float* __restrict__ gw3,
                         const float* __restrict__ mw1, const float* __restrict__ mw2,
                         const float* __restrict__ mw3,
                         _Float16* Wa1, _Float16* Wg1r, _Float16* Wg1c,
                         _Float16* Wg2r, _Float16* Wg2c,
                         float* wg3r, float* wg3c,
                         float* Wm1, _Float16* Wm2, float* wm3) {
  const int tid = threadIdx.x;
  for (int i = tid; i < 64 * 64; i += 256) Wa1[i] = (_Float16)aw1[i];
  // gw1: (64,65,3,3). row variant kh=1, taps over kw; col variant kw=1.
  for (int i = tid; i < 3 * 64 * 96; i += 256) {
    int t = i / (64 * 96); int rem = i % (64 * 96);
    int o = rem / 96;      int k = rem % 96;
    _Float16 vr = (_Float16)0.0f, vc = (_Float16)0.0f;
    if (k < 65) {
      vr = (_Float16)gw1[o * 585 + k * 9 + 3 + t];
      vc = (_Float16)gw1[o * 585 + k * 9 + t * 3 + 1];
    }
    Wg1r[i] = vr; Wg1c[i] = vc;
  }
  // gw2: (64,64,3,3)
  for (int i = tid; i < 3 * 64 * 64; i += 256) {
    int t = i / 4096; int rem = i % 4096;
    int o = rem >> 6; int c = rem & 63;
    Wg2r[i] = (_Float16)gw2[o * 576 + c * 9 + 3 + t];
    Wg2c[i] = (_Float16)gw2[o * 576 + c * 9 + t * 3 + 1];
  }
  // gw3: (1,64,3,3)
  for (int i = tid; i < 3 * 64; i += 256) {
    int t = i >> 6, c = i & 63;
    wg3r[i] = gw3[c * 9 + 3 + t];
    wg3c[i] = gw3[c * 9 + t * 3 + 1];
  }
  // mw1: (64,5,3,3) -> Wm1[o][t][c] f32
  for (int i = tid; i < 64 * 45; i += 256) {
    int o = i / 45; int r2 = i % 45; int t = r2 / 5; int c = r2 % 5;
    Wm1[i] = mw1[o * 45 + c * 9 + t];
  }
  // mw2: (64,64,3,3) -> Wm2[t][o][c] f16
  for (int i = tid; i < 9 * 64 * 64; i += 256) {
    int t = i / 4096; int rem = i % 4096;
    int o = rem >> 6; int c = rem & 63;
    Wm2[i] = (_Float16)mw2[o * 576 + c * 9 + t];
  }
  // mw3: (1,64,3,3) -> wm3[t][c] f32
  for (int i = tid; i < 9 * 64; i += 256) {
    int t = i >> 6, c = i & 63;
    wm3[i] = mw3[c * 9 + t];
  }
}

// ---------------------------------------------------------------------------
// Pack activations: NCHW f32 -> [b][h][w][32] f16 ; data f16 ; init X5 (8ch).
// ---------------------------------------------------------------------------
__global__ void k_pack_x(const float* __restrict__ features,
                         const float* __restrict__ data,
                         _Float16* __restrict__ Xf, _Float16* __restrict__ Df,
                         _Float16* __restrict__ X5) {
  const int idx = blockIdx.x * 256 + threadIdx.x;   // pixel id, 0..262143
  const int b = idx >> 16; const int hw = idx & 65535;
  const float* fsrc = features + (size_t)b * C_ * H_ * W_ + hw;
  _Float16* xdst = Xf + (size_t)idx * 32;
#pragma unroll 4
  for (int c = 0; c < 32; ++c) xdst[c] = (_Float16)fsrc[(size_t)c * H_ * W_];
  const float dv = data[(size_t)b * H_ * W_ + hw];
  Df[idx] = (_Float16)dv;
  _Float16* x5 = X5 + (size_t)idx * 8;
#pragma unroll
  for (int c = 0; c < 8; ++c) x5[c] = (_Float16)0.0f;
  x5[4] = (_Float16)dv;
}

// ---------------------------------------------------------------------------
// Per-direction stage 1: attention (1x1, K=64) + conv1 (3 taps, K=96) WMMA.
// One workgroup = one full 256-pixel line; each wave owns 32 pixels (two
// A tiles per B fragment).
// ---------------------------------------------------------------------------
__global__ void k_dir_attn_conv1(const _Float16* __restrict__ Xf,
                                 const _Float16* __restrict__ Df,
                                 const _Float16* __restrict__ Wa1g,
                                 const _Float16* __restrict__ Wg1g,
                                 const float* __restrict__ ab1,
                                 const float* __restrict__ aw2,
                                 const float* __restrict__ ab2,
                                 const float* __restrict__ gb1,
                                 _Float16* __restrict__ H1,
                                 float* __restrict__ Abuf,
                                 int axis, int s_nf, int s_nbf) {
  __shared__ __align__(16) _Float16 Xt[258 * XT_S];       // 256 px + halo
  __shared__ __align__(16) _Float16 Wa1[64 * OT_S];
  __shared__ __align__(16) _Float16 Wg1[3 * 64 * XT_S];
  const int tid = threadIdx.x;
  const int li  = blockIdx.x;
  const int b   = li / 255;
  const int l   = li % 255;

  // weights -> LDS with padded (conflict-free) strides, row-wise uint4 copies
  for (int i = tid; i < 64; i += 256) {
    const uint4* s = (const uint4*)(Wa1g + i * 64);
    uint4* d = (uint4*)(Wa1 + i * OT_S);
#pragma unroll
    for (int q = 0; q < 8; ++q) d[q] = s[q];
  }
  for (int i = tid; i < 192; i += 256) {
    const uint4* s = (const uint4*)(Wg1g + i * 96);
    uint4* d = (uint4*)(Wg1 + i * XT_S);
#pragma unroll
    for (int q = 0; q < 12; ++q) d[q] = s[q];
  }
  // X tile: 258 rows (256 pixels + halo), 96 ch (nf 32 | nbf 32 | d | 0 pad)
  for (int i = tid; i < 258; i += 256) {
    const int pp = i - 1;
    _Float16* row = Xt + i * XT_S;
    if (pp >= 0 && pp < 256) {
      const int hn = (axis == 0) ? (l + s_nf)  : pp;
      const int wn = (axis == 0) ? pp          : (l + s_nf);
      const int hb = (axis == 0) ? (l + s_nbf) : pp;
      const int wb = (axis == 0) ? pp          : (l + s_nbf);
      const size_t in = (size_t)(b * 256 + hn) * 256 + wn;
      const size_t ib = (size_t)(b * 256 + hb) * 256 + wb;
      const uint4* pn = (const uint4*)(Xf + in * 32);
      const uint4* pb = (const uint4*)(Xf + ib * 32);
      uint4* r0 = (uint4*)row;
#pragma unroll
      for (int q = 0; q < 4; ++q) r0[q] = pn[q];
#pragma unroll
      for (int q = 0; q < 4; ++q) r0[4 + q] = pb[q];
      row[64] = Df[in];
      for (int c = 65; c < 96; ++c) row[c] = (_Float16)0.0f;
    } else {
      uint4 z = {0, 0, 0, 0};
      uint4* r0 = (uint4*)row;
#pragma unroll
      for (int q = 0; q < 12; ++q) r0[q] = z;
    }
  }
  __syncthreads();

  const int wv   = tid >> 5;
  const int lane = tid & 31;
  const int col  = lane & 15;
  const int hi   = (lane >> 4) & 1;
  const int m0   = wv * 32;

  // ---- attention GEMM: [32 px] x [K=64] x [N=64] ----
  v8f ha0[4] = {}, ha1[4] = {};
#pragma unroll
  for (int kk = 0; kk < 2; ++kk) {
    v16h a0 = load_a_frag(Xt, XT_S, m0 + 1, kk * 32);
    v16h a1 = load_a_frag(Xt, XT_S, m0 + 17, kk * 32);
#pragma unroll
    for (int nt = 0; nt < 4; ++nt) {
      v16h bf = load_b_frag(Wa1, OT_S, nt * 16, kk * 32);
      ha0[nt] = wmma_f16(a0, bf, ha0[nt]);
      ha1[nt] = wmma_f16(a1, bf, ha1[nt]);
    }
  }
  // relu(+ab1) . aw2 , reduce across N (cross-lane), sigmoid(+ab2)
  float asum0[8], asum1[8];
#pragma unroll
  for (int r = 0; r < 8; ++r) { asum0[r] = 0.f; asum1[r] = 0.f; }
#pragma unroll
  for (int nt = 0; nt < 4; ++nt) {
    const int o = nt * 16 + col;
    const float w2 = aw2[o];
    const float b1 = ab1[o];
#pragma unroll
    for (int r = 0; r < 8; ++r) {
      float v0 = ha0[nt][r] + b1; v0 = v0 > 0.f ? v0 : 0.f;
      float v1 = ha1[nt][r] + b1; v1 = v1 > 0.f ? v1 : 0.f;
      asum0[r] += v0 * w2;
      asum1[r] += v1 * w2;
    }
  }
  const float a_b2 = ab2[0];
#pragma unroll
  for (int r = 0; r < 8; ++r) {
    float t0 = asum0[r], t1 = asum1[r];
    t0 += __shfl_xor(t0, 1); t1 += __shfl_xor(t1, 1);
    t0 += __shfl_xor(t0, 2); t1 += __shfl_xor(t1, 2);
    t0 += __shfl_xor(t0, 4); t1 += __shfl_xor(t1, 4);
    t0 += __shfl_xor(t0, 8); t1 += __shfl_xor(t1, 8);
    if (col == 0) {
      const int p = m0 + r + hi * 8;
      Abuf[(size_t)li * 256 + p]      = 1.f / (1.f + __expf(-(t0 + a_b2)));
      Abuf[(size_t)li * 256 + p + 16] = 1.f / (1.f + __expf(-(t1 + a_b2)));
    }
  }

  // ---- conv1 GEMM: 3 taps x K=96 (65 real + pad) x N=64 ----
  v8f c0[4] = {}, c1[4] = {};
#pragma unroll
  for (int t3 = 0; t3 < 3; ++t3) {
#pragma unroll
    for (int kk = 0; kk < 3; ++kk) {
      v16h a0 = load_a_frag(Xt, XT_S, m0 + t3, kk * 32);
      v16h a1 = load_a_frag(Xt, XT_S, m0 + 16 + t3, kk * 32);
#pragma unroll
      for (int nt = 0; nt < 4; ++nt) {
        v16h bf = load_b_frag(Wg1 + t3 * 64 * XT_S, XT_S, nt * 16, kk * 32);
        c0[nt] = wmma_f16(a0, bf, c0[nt]);
        c1[nt] = wmma_f16(a1, bf, c1[nt]);
      }
    }
  }
#pragma unroll
  for (int nt = 0; nt < 4; ++nt) {
    const int o = nt * 16 + col;
    const float bia = gb1[o];
#pragma unroll
    for (int r = 0; r < 8; ++r) {
      const int p = m0 + r + hi * 8;
      float v0 = c0[nt][r] + bia; v0 = v0 > 0.f ? v0 : 0.f;
      float v1 = c1[nt][r] + bia; v1 = v1 > 0.f ? v1 : 0.f;
      H1[((size_t)li * 256 + p) * 64 + o]        = (_Float16)v0;
      H1[((size_t)li * 256 + p + 16) * 64 + o]   = (_Float16)v1;
    }
  }
}

// ---------------------------------------------------------------------------
// Per-direction stage 2: conv2 (3 taps, K=64) WMMA, 32 px per wave.
// ---------------------------------------------------------------------------
__global__ void k_dir_conv2(const _Float16* __restrict__ H1,
                            const _Float16* __restrict__ Wg2g,
                            const float* __restrict__ gb2,
                            _Float16* __restrict__ H2) {
  __shared__ __align__(16) _Float16 Ot[258 * OT_S];
  __shared__ __align__(16) _Float16 Wg2[3 * 64 * OT_S];
  const int tid = threadIdx.x;
  const int li  = blockIdx.x;

  for (int i = tid; i < 192; i += 256) {
    const uint4* s = (const uint4*)(Wg2g + i * 64);
    uint4* d = (uint4*)(Wg2 + i * OT_S);
#pragma unroll
    for (int q = 0; q < 8; ++q) d[q] = s[q];
  }
  for (int i = tid; i < 258; i += 256) {
    const int pp = i - 1;
    uint4* d = (uint4*)(Ot + i * OT_S);
    if (pp >= 0 && pp < 256) {
      const uint4* s = (const uint4*)(H1 + ((size_t)li * 256 + pp) * 64);
#pragma unroll
      for (int q = 0; q < 8; ++q) d[q] = s[q];
      // speculative prefetch of the next line's row (global_prefetch_b8)
      if (li + 1 < (int)gridDim.x)
        __builtin_prefetch(H1 + ((size_t)(li + 1) * 256 + pp) * 64, 0, 1);
    } else {
      uint4 z = {0, 0, 0, 0};
#pragma unroll
      for (int q = 0; q < 8; ++q) d[q] = z;
    }
  }
  __syncthreads();

  const int wv   = tid >> 5;
  const int lane = tid & 31;
  const int col  = lane & 15;
  const int hi   = (lane >> 4) & 1;
  const int m0   = wv * 32;

  v8f c0[4] = {}, c1[4] = {};
#pragma unroll
  for (int t3 = 0; t3 < 3; ++t3) {
#pragma unroll
    for (int kk = 0; kk < 2; ++kk) {
      v16h a0 = load_a_frag(Ot, OT_S, m0 + t3, kk * 32);
      v16h a1 = load_a_frag(Ot, OT_S, m0 + 16 + t3, kk * 32);
#pragma unroll
      for (int nt = 0; nt < 4; ++nt) {
        v16h bf = load_b_frag(Wg2 + t3 * 64 * OT_S, OT_S, nt * 16, kk * 32);
        c0[nt] = wmma_f16(a0, bf, c0[nt]);
        c1[nt] = wmma_f16(a1, bf, c1[nt]);
      }
    }
  }
#pragma unroll
  for (int nt = 0; nt < 4; ++nt) {
    const int o = nt * 16 + col;
    const float bia = gb2[o];
#pragma unroll
    for (int r = 0; r < 8; ++r) {
      const int p = m0 + r + hi * 8;
      float v0 = c0[nt][r] + bia; v0 = v0 > 0.f ? v0 : 0.f;
      float v1 = c1[nt][r] + bia; v1 = v1 > 0.f ? v1 : 0.f;
      H2[((size_t)li * 256 + p) * 64 + o]      = (_Float16)v0;
      H2[((size_t)li * 256 + p + 16) * 64 + o] = (_Float16)v1;
    }
  }
}

// ---------------------------------------------------------------------------
// Per-direction stage 3: conv3 (64->1, 3 taps) * sigmoid gate, scatter to X5.
// ---------------------------------------------------------------------------
__global__ void k_dir_msg(const _Float16* __restrict__ H2,
                          const float* __restrict__ wg3,
                          const float* __restrict__ gb3,
                          const float* __restrict__ Abuf,
                          _Float16* __restrict__ X5,
                          int axis, int s_out, int ch) {
  const int idx = blockIdx.x * 256 + threadIdx.x;   // li*256 + p
  const int li = idx >> 8; const int p = idx & 255;
  const int b = li / 255;  const int l = li % 255;
  float acc = gb3[0];
#pragma unroll
  for (int t = 0; t < 3; ++t) {
    const int pp = p + t - 1;
    if (pp < 0 || pp > 255) continue;
    const _Float16* src = H2 + ((size_t)li * 256 + pp) * 64;
    const float* wt = wg3 + t * 64;
    float s = 0.f;
#pragma unroll 8
    for (int c = 0; c < 64; ++c) s += (float)src[c] * wt[c];
    acc += s;
  }
  acc *= Abuf[idx];
  const int ho = (axis == 0) ? (l + s_out) : p;
  const int wo = (axis == 0) ? p : (l + s_out);
  X5[((size_t)(b * 256 + ho) * 256 + wo) * 8 + ch] = (_Float16)acc;
}

// ---------------------------------------------------------------------------
// Fusion net conv1: 5->64, 3x3 (tiny FLOPs: VALU).
// ---------------------------------------------------------------------------
__global__ void k_final1(const _Float16* __restrict__ X5,
                         const float* __restrict__ Wm1,
                         const float* __restrict__ mb1,
                         _Float16* __restrict__ H3) {
  __shared__ float Wl[64 * 45];
  for (int i = threadIdx.x; i < 64 * 45; i += 256) Wl[i] = Wm1[i];
  __syncthreads();
  const int idx = blockIdx.x * 256 + threadIdx.x;
  const int b = idx >> 16; const int hw = idx & 65535;
  const int h = hw >> 8;   const int w = hw & 255;
  float xv[45];
#pragma unroll
  for (int t = 0; t < 9; ++t) {
    const int hh = h + t / 3 - 1, ww = w + t % 3 - 1;
    const bool ok = (hh >= 0 && hh < 256 && ww >= 0 && ww < 256);
    const _Float16* s = X5 + ((size_t)(b * 256 + hh) * 256 + ww) * 8;
#pragma unroll
    for (int c = 0; c < 5; ++c) xv[t * 5 + c] = ok ? (float)s[c] : 0.f;
  }
  _Float16* dst = H3 + (size_t)idx * 64;
  for (int o = 0; o < 64; ++o) {
    float s = mb1[o];
    const float* wr = Wl + o * 45;
#pragma unroll
    for (int j = 0; j < 45; ++j) s += xv[j] * wr[j];
    dst[o] = (_Float16)(s > 0.f ? s : 0.f);
  }
}

// ---------------------------------------------------------------------------
// Fusion net conv2: 64->64, 3x3 == 9-tap x K=64 WMMA GEMM (dominant GEMM).
// ---------------------------------------------------------------------------
__global__ void k_final2(const _Float16* __restrict__ H3,
                         const _Float16* __restrict__ Wm2g,
                         const float* __restrict__ mb2,
                         _Float16* __restrict__ H4) {
  __shared__ __align__(16) _Float16 Ht[3 * 130 * OT_S];
  __shared__ __align__(16) _Float16 Wm2[9 * 64 * OT_S];
  const int tid = threadIdx.x;
  const int li  = blockIdx.x >> 1;
  const int p0  = (blockIdx.x & 1) * 128;
  const int b   = li >> 8;
  const int h   = li & 255;

  for (int i = tid; i < 576; i += 256) {
    const uint4* s = (const uint4*)(Wm2g + i * 64);
    uint4* d = (uint4*)(Wm2 + i * OT_S);
#pragma unroll
    for (int q = 0; q < 8; ++q) d[q] = s[q];
  }
  for (int i = tid; i < 3 * 130; i += 256) {
    const int rr = i / 130, ii = i % 130;
    const int hh = h + rr - 1, pp = p0 - 1 + ii;
    uint4* d = (uint4*)(Ht + i * OT_S);
    if (hh >= 0 && hh < 256 && pp >= 0 && pp < 256) {
      const uint4* s = (const uint4*)(H3 + ((size_t)(b * 256 + hh) * 256 + pp) * 64);
#pragma unroll
      for (int q = 0; q < 8; ++q) d[q] = s[q];
    } else {
      uint4 z = {0, 0, 0, 0};
#pragma unroll
      for (int q = 0; q < 8; ++q) d[q] = z;
    }
  }
  __syncthreads();

  const int wv   = tid >> 5;
  const int lane = tid & 31;
  const int col  = lane & 15;
  const int hi   = (lane >> 4) & 1;
  const int m0   = wv * 16;

  v8f acc[4] = {};
#pragma unroll
  for (int th = 0; th < 3; ++th) {
#pragma unroll
    for (int tw = 0; tw < 3; ++tw) {
      const int t = th * 3 + tw;
#pragma unroll
      for (int kk = 0; kk < 2; ++kk) {
        v16h a = load_a_frag(Ht + th * 130 * OT_S, OT_S, m0 + tw, kk * 32);
#pragma unroll
        for (int nt = 0; nt < 4; ++nt) {
          v16h bf = load_b_frag(Wm2 + t * 64 * OT_S, OT_S, nt * 16, kk * 32);
          acc[nt] = wmma_f16(a, bf, acc[nt]);
        }
      }
    }
  }
#pragma unroll
  for (int nt = 0; nt < 4; ++nt) {
    const int o = nt * 16 + col;
    const float bia = mb2[o];
#pragma unroll
    for (int r = 0; r < 8; ++r) {
      float v = acc[nt][r] + bia;
      v = v > 0.f ? v : 0.f;
      const int p = p0 + m0 + r + hi * 8;
      H4[((size_t)li * 256 + p) * 64 + o] = (_Float16)v;
    }
  }
}

// ---------------------------------------------------------------------------
// Fusion net conv3: 64->1, 3x3 (VALU) -> f32 belief.
// ---------------------------------------------------------------------------
__global__ void k_final3(const _Float16* __restrict__ H4,
                         const float* __restrict__ wm3,
                         const float* __restrict__ mb3,
                         float* __restrict__ out) {
  const int idx = blockIdx.x * 256 + threadIdx.x;
  const int b = idx >> 16; const int hw = idx & 65535;
  const int h = hw >> 8;   const int w = hw & 255;
  float acc = mb3[0];
#pragma unroll
  for (int t = 0; t < 9; ++t) {
    const int hh = h + t / 3 - 1, ww = w + t % 3 - 1;
    if (hh < 0 || hh > 255 || ww < 0 || ww > 255) continue;
    const _Float16* s = H4 + ((size_t)(b * 256 + hh) * 256 + ww) * 64;
    const float* wt = wm3 + t * 64;
    float ss = 0.f;
#pragma unroll 8
    for (int c = 0; c < 64; ++c) ss += (float)s[c] * wt[c];
    acc += ss;
  }
  out[idx] = acc;
}

// ---------------------------------------------------------------------------
extern "C" void kernel_launch(void* const* d_in, const int* in_sizes, int n_in,
                              void* d_out, int out_size, void* d_ws, size_t ws_size,
                              hipStream_t stream) {
  (void)in_sizes; (void)n_in; (void)out_size; (void)ws_size;
  const float* features = (const float*)d_in[0];
  const float* data     = (const float*)d_in[1];
  const float* aw1 = (const float*)d_in[3];
  const float* ab1 = (const float*)d_in[4];
  const float* aw2 = (const float*)d_in[5];
  const float* ab2 = (const float*)d_in[6];
  const float* gw1 = (const float*)d_in[7];
  const float* gb1 = (const float*)d_in[8];
  const float* gw2 = (const float*)d_in[9];
  const float* gb2 = (const float*)d_in[10];
  const float* gw3 = (const float*)d_in[11];
  const float* gb3 = (const float*)d_in[12];
  const float* mw1 = (const float*)d_in[13];
  const float* mb1 = (const float*)d_in[14];
  const float* mw2 = (const float*)d_in[15];
  const float* mb2 = (const float*)d_in[16];
  const float* mw3 = (const float*)d_in[17];
  const float* mb3 = (const float*)d_in[18];

  char* ws = (char*)d_ws;
  size_t off = 0;
  auto alloc = [&](size_t bytes) -> void* {
    void* p = ws + off;
    off += (bytes + 255) & ~(size_t)255;
    return p;
  };
  _Float16* Xf   = (_Float16*)alloc((size_t)B_ * H_ * W_ * 32 * 2);
  _Float16* Df   = (_Float16*)alloc((size_t)B_ * H_ * W_ * 2);
  _Float16* X5   = (_Float16*)alloc((size_t)B_ * H_ * W_ * 8 * 2);
  _Float16* H1   = (_Float16*)alloc((size_t)B_ * H_ * W_ * 64 * 2);  // also H3
  _Float16* H2   = (_Float16*)alloc((size_t)B_ * H_ * W_ * 64 * 2);  // also H4
  float*    Abuf = (float*)alloc((size_t)NLINES_ * 256 * 4);
  _Float16* Wa1p = (_Float16*)alloc(64 * 64 * 2);
  _Float16* Wg1r = (_Float16*)alloc(3 * 64 * 96 * 2);
  _Float16* Wg1c = (_Float16*)alloc(3 * 64 * 96 * 2);
  _Float16* Wg2r = (_Float16*)alloc(3 * 64 * 64 * 2);
  _Float16* Wg2c = (_Float16*)alloc(3 * 64 * 64 * 2);
  float*    wg3r = (float*)alloc(3 * 64 * 4);
  float*    wg3c = (float*)alloc(3 * 64 * 4);
  float*    Wm1p = (float*)alloc(64 * 45 * 4);
  _Float16* Wm2p = (_Float16*)alloc(9 * 64 * 64 * 2);
  float*    wm3p = (float*)alloc(9 * 64 * 4);

  k_pack_w<<<1, 256, 0, stream>>>(aw1, gw1, gw2, gw3, mw1, mw2, mw3,
                                  Wa1p, Wg1r, Wg1c, Wg2r, Wg2c, wg3r, wg3c,
                                  Wm1p, Wm2p, wm3p);
  k_pack_x<<<(B_ * H_ * W_) / 256, 256, 0, stream>>>(features, data, Xf, Df, X5);

  struct Dir { int axis, s_nf, s_nbf, s_out, ch;
               const _Float16* w1; const _Float16* w2; const float* w3; };
  const Dir dirs[4] = {
    {0, 1, 0, 0, 0, Wg1r, Wg2r, wg3r},   // up
    {0, 0, 1, 1, 1, Wg1r, Wg2r, wg3r},   // down
    {1, 1, 0, 0, 2, Wg1c, Wg2c, wg3c},   // left
    {1, 0, 1, 1, 3, Wg1c, Wg2c, wg3c},   // right
  };
  for (int d = 0; d < 4; ++d) {
    k_dir_attn_conv1<<<NLINES_, 256, 0, stream>>>(
        Xf, Df, Wa1p, dirs[d].w1, ab1, aw2, ab2, gb1, H1, Abuf,
        dirs[d].axis, dirs[d].s_nf, dirs[d].s_nbf);
    k_dir_conv2<<<NLINES_, 256, 0, stream>>>(H1, dirs[d].w2, gb2, H2);
    k_dir_msg<<<NLINES_, 256, 0, stream>>>(H2, dirs[d].w3, gb3, Abuf, X5,
                                           dirs[d].axis, dirs[d].s_out, dirs[d].ch);
  }
  k_final1<<<(B_ * H_ * W_) / 256, 256, 0, stream>>>(X5, Wm1p, mb1, H1);
  k_final2<<<(B_ * H_) * 2, 256, 0, stream>>>(H1, Wm2p, mb2, H2);
  k_final3<<<(B_ * H_ * W_) / 256, 256, 0, stream>>>(H2, wm3p, mb3, (float*)d_out);
}
